// CoRoPE_86964497809857
// MI455X (gfx1250) — matrix operand order repaired and checked by reading, not compiled, not run on hardware
//
#include <hip/hip_runtime.h>
#include <stdint.h>

#define B_  4
#define T_  2048
#define E_  2048
#define H_  16
#define KV_ 4
#define D_  128

typedef __attribute__((ext_vector_type(16))) __bf16       v16bf;
typedef __attribute__((ext_vector_type(8)))  float        v8f;
typedef __attribute__((ext_vector_type(8)))  uint32_t     v8u;
typedef __attribute__((ext_vector_type(4)))  unsigned int u32x4;
typedef __attribute__((ext_vector_type(8)))  int          i32x8;
typedef __attribute__((ext_vector_type(4)))  int          i32x4;

#if __has_builtin(__builtin_amdgcn_tensor_load_to_lds)
#define USE_TDM 1
#else
#define USE_TDM 0
#endif

__device__ __forceinline__ uint16_t f32_to_bf16(float f) {
  uint32_t u = __builtin_bit_cast(uint32_t, f);
  u += 0x7fffu + ((u >> 16) & 1u);   // round-to-nearest-even
  return (uint16_t)(u >> 16);
}

// gfx1250 async global->LDS copy (ASYNCcnt path, CDNA5 ISA)
__device__ __forceinline__ void async_copy_b128(uint32_t lds_off, uint64_t gaddr) {
  asm volatile("global_load_async_to_lds_b128 %0, %1, off"
               :: "v"(lds_off), "v"(gaddr) : "memory");
}
__device__ __forceinline__ void wait_async() {
#if __has_builtin(__builtin_amdgcn_s_wait_asynccnt)
  __builtin_amdgcn_s_wait_asynccnt(0);
#else
  asm volatile("s_wait_asynccnt 0x0" ::: "memory");
#endif
}

// ---------------------------------------------------------------------------
// fp32 -> bf16 elementwise convert
// ---------------------------------------------------------------------------
__global__ __launch_bounds__(256) void cvt_f32_bf16(const float* __restrict__ src,
                                                    uint16_t* __restrict__ dst, int n) {
  int i = blockIdx.x * blockDim.x + threadIdx.x;
  int stride = gridDim.x * blockDim.x;
  for (; i < n; i += stride) dst[i] = f32_to_bf16(src[i]);
}

// ---------------------------------------------------------------------------
// bf16 WMMA GEMM: C[MxN] = A[MxK] * B[KxN], row-major.
// Block tile 128x128, BK=32. 8 waves as 4(M)x2(N): wave tile 32x64
// (2 A-frags x 4 B-frags -> 8 WMMAs per K-step). A staged via async-to-LDS,
// B staged transposed [n][k] with key-pair-packed b32 stores.
// ---------------------------------------------------------------------------
template <bool STORE_F32>
__global__ __launch_bounds__(256) void gemm_bf16(const uint16_t* __restrict__ A,
                                                 const uint16_t* __restrict__ Bm,
                                                 void* __restrict__ C,
                                                 int M, int N, int K) {
  __shared__ uint16_t ldsA[128 * 32];   // [m][k]
  __shared__ uint16_t ldsBT[128 * 32];  // [n][k]

  const int tid  = threadIdx.x;
  const int wave = tid >> 5;
  const int lane = tid & 31;
  const int lm   = lane & 15;
  const int half = lane >> 4;
  const int wr   = wave & 3;   // M sub-block (32 rows)
  const int wc   = wave >> 2;  // N sub-block (64 cols)

  const int mBlk = blockIdx.y * 128;
  const int nBlk = blockIdx.x * 128;

  v8f acc[2][4];
#pragma unroll
  for (int i = 0; i < 2; ++i)
#pragma unroll
    for (int c = 0; c < 4; ++c) acc[i][c] = 0.f;

  const int arow = tid >> 1;          // 0..127
  const int acol = (tid & 1) * 16;    // 0,16
  const int bkp  = (tid & 15) * 2;    // k pair 0..30
  const int bn0  = (tid >> 4) * 8;    // 0..120

  for (int kt = 0; kt < K; kt += 32) {
    // stage A tile 128x32 via async copy (2 x b128 per thread)
    {
      uint64_t ga = (uint64_t)(uintptr_t)(A + (size_t)(mBlk + arow) * K + kt + acol);
      uint32_t la = (uint32_t)(uintptr_t)(ldsA + arow * 32 + acol);
      async_copy_b128(la, ga);
      async_copy_b128(la + 16u, ga + 16u);
    }
    // stage B tile 32x128 transposed into [n][k], packing 2 k's per b32 store
    {
      union { uint4 q; uint16_t s[8]; } r0, r1;
      const uint16_t* bp = Bm + (size_t)(kt + bkp) * N + nBlk + bn0;
      r0.q = *(const uint4*)bp;
      r1.q = *(const uint4*)(bp + N);
#pragma unroll
      for (int j = 0; j < 8; ++j)
        *(uint32_t*)(ldsBT + (bn0 + j) * 32 + bkp) =
            (uint32_t)r0.s[j] | ((uint32_t)r1.s[j] << 16);
    }
    if (kt + 32 < K) {  // prefetch next tiles (global_prefetch_b8)
      __builtin_prefetch(A + (size_t)(mBlk + arow) * K + kt + 32 + acol, 0, 3);
      __builtin_prefetch(Bm + (size_t)(kt + 32 + bkp) * N + nBlk + bn0, 0, 3);
    }
    wait_async();
    __syncthreads();

    // A fragments for rows wr*32 .. wr*32+31
    v16bf af[2];
#pragma unroll
    for (int i = 0; i < 2; ++i) {
      v8u au;
#pragma unroll
      for (int v = 0; v < 8; ++v) {
        int k = (half << 4) + (v << 1);
        au[v] = *(const uint32_t*)(ldsA + (wr * 32 + i * 16 + lm) * 32 + k);
      }
      af[i] = __builtin_bit_cast(v16bf, au);
    }

#pragma unroll
    for (int c = 0; c < 4; ++c) {
      v8u bu;
#pragma unroll
      for (int v = 0; v < 8; ++v) {
        int k = (half << 4) + (v << 1);
        bu[v] = *(const uint32_t*)(ldsBT + (wc * 64 + c * 16 + lm) * 32 + k);
      }
      v16bf bf = __builtin_bit_cast(v16bf, bu);
      acc[0][c] = __builtin_amdgcn_wmma_f32_16x16x32_bf16(
          false, af[0], false, bf, (short)0, acc[0][c], false, false);
      acc[1][c] = __builtin_amdgcn_wmma_f32_16x16x32_bf16(
          false, af[1], false, bf, (short)0, acc[1][c], false, false);
    }
    __syncthreads();
  }

  // epilogue: C-layout element (r,lane) -> row r+8*half, col lm
#pragma unroll
  for (int i = 0; i < 2; ++i) {
#pragma unroll
    for (int c = 0; c < 4; ++c) {
#pragma unroll
      for (int r = 0; r < 8; ++r) {
        size_t row = (size_t)(mBlk + wr * 32 + i * 16 + r + (half << 3));
        size_t col = (size_t)(nBlk + wc * 64 + c * 16 + lm);
        float val = acc[i][c][r];
        if (STORE_F32) ((float*)C)[row * N + col] = val;
        else           ((uint16_t*)C)[row * N + col] = f32_to_bf16(val);
      }
    }
  }
}

// ---------------------------------------------------------------------------
// Fused causal GQA flash attention (bf16 WMMA, fp32 accum).
// Grid: (T/64, B*H). 128 threads = 4 waves; wave w owns query rows
// qb+16w .. qb+16w+15. Iterates 32-key tiles with online softmax.
// K tile staged by the Tensor Data Mover when available; waves skip
// fully-masked tiles (barriers stay uniform).
// ---------------------------------------------------------------------------
__global__ __launch_bounds__(128) void flash_attn_gqa(const uint16_t* __restrict__ Q,
                                                      const uint16_t* __restrict__ Kb,
                                                      const uint16_t* __restrict__ Vb,
                                                      uint16_t* __restrict__ Y) {
  __shared__ uint16_t ldsK[32 * 128];    // [key][d]
  __shared__ uint16_t ldsVT[128 * 32];   // [d][key]
  __shared__ uint16_t ldsP[4][16 * 32];  // per-wave P scratch [m][n]

  const int tid  = threadIdx.x;
  const int wave = tid >> 5;
  const int lane = tid & 31;
  const int lm   = lane & 15;
  const int half = lane >> 4;

  const int bh = blockIdx.y;
  const int b  = bh >> 4;        // batch
  const int hq = bh & 15;        // query head
  const int kv = hq >> 2;        // kv head = hq / G
  const int qb = blockIdx.x * 64;
  const int rowLo = qb + wave * 16;

  const float scale = 0.08838834764831845f;  // 1/sqrt(128)

  // Q fragments for the wave's 16 rows, full D=128 (4 chunks of K=32)
  v16bf qf[4];
  {
    const int qrow = rowLo + lm;
    const uint16_t* qbase = Q + ((size_t)(b * T_ + qrow) * H_ + hq) * D_;
#pragma unroll
    for (int c = 0; c < 4; ++c) {
      v8u qu;
#pragma unroll
      for (int v = 0; v < 8; ++v) {
        int k = (half << 4) + (v << 1);
        qu[v] = *(const uint32_t*)(qbase + c * 32 + k);
      }
      qf[c] = __builtin_bit_cast(v16bf, qu);
    }
  }

  float mrow[8], lrow[8];
#pragma unroll
  for (int r = 0; r < 8; ++r) { mrow[r] = -__builtin_inff(); lrow[r] = 0.f; }
  v8f oacc[8];
#pragma unroll
  for (int c = 0; c < 8; ++c) oacc[c] = 0.f;

  const int nkt = (qb >> 5) + 2;   // causal: key tiles covering [0, qb+64)

  for (int kt = 0; kt < nkt; ++kt) {
    const int kt0 = kt * 32;

    // ---- stage K tile 32x128 ----
#if USE_TDM
    if (wave == 0) {
      uint64_t gk = (uint64_t)(uintptr_t)(Kb + ((size_t)(b * T_ + kt0) * KV_ + kv) * D_);
      u32x4 g0;
      g0[0] = 1u;                                            // count=1 (valid user D#)
      g0[1] = (uint32_t)(uintptr_t)(&ldsK[0]);               // lds_addr
      g0[2] = (uint32_t)gk;                                  // global_addr[31:0]
      g0[3] = ((uint32_t)(gk >> 32) & 0x01FFFFFFu) | 0x80000000u;  // addr[56:32] | type=2
      i32x8 g1 = 0;
      g1[0] = (int)(1u << 16);                               // data_size = 2 bytes
      g1[1] = (int)(128u << 16);                             // tensor_dim0 = 128
      g1[3] = (int)((128u << 16) | 0x10u);                   // tile_dim0=128 | tensor_dim1 hi
      g1[4] = 32;                                            // tile_dim1 = 32
      g1[5] = KV_ * D_;                                      // tensor_dim0_stride = 512
      i32x4 g2 = 0, g3 = 0;
      i32x8 g4 = 0;                                          // unused trailing group (6-arg form)
      __builtin_amdgcn_tensor_load_to_lds(g0, g1, g2, g3, g4, 0);
    }
#else
    {
      const int krow  = tid >> 2;
      const int kcol0 = (tid & 3) * 32;
      const uint4* src = (const uint4*)(Kb + ((size_t)(b * T_ + kt0 + krow) * KV_ + kv) * D_ + kcol0);
      uint4* dst = (uint4*)(ldsK + krow * 128 + kcol0);
#pragma unroll
      for (int j = 0; j < 4; ++j) dst[j] = src[j];
    }
#endif
    // ---- stage V tile transposed -> [d][key], 2 keys packed per b32 ----
    {
      const int kp = (tid & 15) * 2;   // key pair base
      const int d0 = (tid >> 4) * 16;  // d chunk
      union { uint4 q[2]; uint16_t s[16]; } va, vb2;
      const uint16_t* v0 = Vb + ((size_t)(b * T_ + kt0 + kp) * KV_ + kv) * D_ + d0;
      const uint16_t* v1 = v0 + (size_t)KV_ * D_;
      va.q[0]  = ((const uint4*)v0)[0]; va.q[1]  = ((const uint4*)v0)[1];
      vb2.q[0] = ((const uint4*)v1)[0]; vb2.q[1] = ((const uint4*)v1)[1];
#pragma unroll
      for (int j = 0; j < 16; ++j)
        *(uint32_t*)(ldsVT + (d0 + j) * 32 + kp) =
            (uint32_t)va.s[j] | ((uint32_t)vb2.s[j] << 16);
    }
#if USE_TDM
    if (wave == 0) __builtin_amdgcn_s_wait_tensorcnt(0);
#endif
    __syncthreads();

    // wave-uniform tile classification (barriers stay outside)
    const bool active   = (kt0 <= rowLo + 15);  // else fully masked: skip
    const bool needMask = (kt0 + 31 > rowLo);   // diagonal tile

    if (active) {
      // S = Q K^T for 16 rows x 32 keys (two N-chunks), accumulate over D
      v8f s0 = 0.f, s1 = 0.f;
#pragma unroll
      for (int c = 0; c < 4; ++c) {
        v8u b0, b1;
#pragma unroll
        for (int v = 0; v < 8; ++v) {
          int kd = (half << 4) + (v << 1);
          b0[v] = *(const uint32_t*)(ldsK + lm * 128 + c * 32 + kd);
          b1[v] = *(const uint32_t*)(ldsK + (16 + lm) * 128 + c * 32 + kd);
        }
        s0 = __builtin_amdgcn_wmma_f32_16x16x32_bf16(
            false, qf[c], false, __builtin_bit_cast(v16bf, b0), (short)0, s0, false, false);
        s1 = __builtin_amdgcn_wmma_f32_16x16x32_bf16(
            false, qf[c], false, __builtin_bit_cast(v16bf, b1), (short)0, s1, false, false);
      }

      // online softmax (row m = r + 8*half spans the 16 lanes of this half)
#pragma unroll
      for (int r = 0; r < 8; ++r) {
        const int qi = rowLo + r + (half << 3);
        float a0 = s0[r] * scale;
        float a1 = s1[r] * scale;
        if (needMask) {
          if (kt0 + lm > qi)      a0 = -__builtin_inff();
          if (kt0 + 16 + lm > qi) a1 = -__builtin_inff();
        }
        float rmax = fmaxf(a0, a1);
#pragma unroll
        for (int off = 8; off >= 1; off >>= 1)
          rmax = fmaxf(rmax, __shfl_xor(rmax, off, 16));
        float mnew = fmaxf(mrow[r], rmax);
        float al   = __expf(mrow[r] - mnew);
        float p0   = __expf(a0 - mnew);
        float p1   = __expf(a1 - mnew);
        float rsum = p0 + p1;
#pragma unroll
        for (int off = 8; off >= 1; off >>= 1)
          rsum += __shfl_xor(rsum, off, 16);
        lrow[r] = lrow[r] * al + rsum;
        mrow[r] = mnew;
#pragma unroll
        for (int c = 0; c < 8; ++c) oacc[c][r] *= al;
        const int mm = r + (half << 3);
        ldsP[wave][mm * 32 + lm]      = f32_to_bf16(p0);
        ldsP[wave][mm * 32 + 16 + lm] = f32_to_bf16(p1);
      }
    }
    __syncthreads();

    if (active) {
      // P accumulator-layout -> A-fragment re-layout via LDS
      v8u pu;
#pragma unroll
      for (int v = 0; v < 8; ++v) {
        int k = (half << 4) + (v << 1);
        pu[v] = *(const uint32_t*)(&ldsP[wave][lm * 32 + k]);
      }
      v16bf pf = __builtin_bit_cast(v16bf, pu);

      // O += P (16x32) x V (32x128): 8 d-chunks
#pragma unroll
      for (int c = 0; c < 8; ++c) {
        v8u bv;
#pragma unroll
        for (int v = 0; v < 8; ++v) {
          int key = (half << 4) + (v << 1);
          bv[v] = *(const uint32_t*)(ldsVT + (c * 16 + lm) * 32 + key);
        }
        oacc[c] = __builtin_amdgcn_wmma_f32_16x16x32_bf16(
            false, pf, false, __builtin_bit_cast(v16bf, bv), (short)0, oacc[c], false, false);
      }
    }
    __syncthreads();
  }

  // normalize and store bf16 Y laid out (B,T,H,D) so Y @ Wo is a plain GEMM
#pragma unroll
  for (int c = 0; c < 8; ++c) {
#pragma unroll
    for (int r = 0; r < 8; ++r) {
      float val = oacc[c][r] / lrow[r];
      size_t t  = (size_t)(rowLo + r + (half << 3));
      Y[((size_t)(b * T_) + t) * (H_ * D_) + hq * D_ + c * 16 + lm] = f32_to_bf16(val);
    }
  }
}

// ---------------------------------------------------------------------------
// launcher
// ---------------------------------------------------------------------------
extern "C" void kernel_launch(void* const* d_in, const int* in_sizes, int n_in,
                              void* d_out, int out_size, void* d_ws, size_t ws_size,
                              hipStream_t stream) {
  (void)in_sizes; (void)n_in; (void)out_size; (void)ws_size;
  const float* x  = (const float*)d_in[0];
  const float* Wq = (const float*)d_in[1];
  const float* Wk = (const float*)d_in[2];
  const float* Wv = (const float*)d_in[3];
  const float* Wo = (const float*)d_in[4];
  float* out = (float*)d_out;

  char* ws = (char*)d_ws;
  size_t off = 0;
  auto alloc = [&](size_t bytes) {
    char* p = ws + off;
    off += (bytes + 255) & ~(size_t)255;
    return p;
  };
  uint16_t* xb   = (uint16_t*)alloc((size_t)B_ * T_ * E_ * 2);
  uint16_t* wqb  = (uint16_t*)alloc((size_t)E_ * H_ * D_ * 2);
  uint16_t* wkb  = (uint16_t*)alloc((size_t)E_ * KV_ * D_ * 2);
  uint16_t* wvb  = (uint16_t*)alloc((size_t)E_ * KV_ * D_ * 2);
  uint16_t* wob  = (uint16_t*)alloc((size_t)H_ * D_ * E_ * 2);
  uint16_t* qbuf = (uint16_t*)alloc((size_t)B_ * T_ * H_ * D_ * 2);
  uint16_t* kbuf = (uint16_t*)alloc((size_t)B_ * T_ * KV_ * D_ * 2);
  uint16_t* vbuf = (uint16_t*)alloc((size_t)B_ * T_ * KV_ * D_ * 2);
  uint16_t* ybuf = (uint16_t*)alloc((size_t)B_ * T_ * H_ * D_ * 2);

  // fp32 -> bf16
  cvt_f32_bf16<<<2048, 256, 0, stream>>>(x,  xb,  B_ * T_ * E_);
  cvt_f32_bf16<<<1024, 256, 0, stream>>>(Wq, wqb, E_ * H_ * D_);
  cvt_f32_bf16<<<512,  256, 0, stream>>>(Wk, wkb, E_ * KV_ * D_);
  cvt_f32_bf16<<<512,  256, 0, stream>>>(Wv, wvb, E_ * KV_ * D_);
  cvt_f32_bf16<<<1024, 256, 0, stream>>>(Wo, wob, H_ * D_ * E_);

  const int M = B_ * T_;  // 8192
  // Q/K/V projections (bf16 outputs)
  gemm_bf16<false><<<dim3((H_ * D_) / 128, M / 128), 256, 0, stream>>>(xb, wqb, qbuf, M, H_ * D_, E_);
  gemm_bf16<false><<<dim3((KV_ * D_) / 128, M / 128), 256, 0, stream>>>(xb, wkb, kbuf, M, KV_ * D_, E_);
  gemm_bf16<false><<<dim3((KV_ * D_) / 128, M / 128), 256, 0, stream>>>(xb, wvb, vbuf, M, KV_ * D_, E_);

  // fused causal GQA flash attention
  flash_attn_gqa<<<dim3(T_ / 64, B_ * H_), 128, 0, stream>>>(qbuf, kbuf, vbuf, ybuf);

  // output projection (fp32 output)
  gemm_bf16<true><<<dim3(E_ / 128, M / 128), 256, 0, stream>>>(ybuf, wob, out, M, E_, H_ * D_);
}
